// Xerxes2DecoderLayer_601295421791
// MI455X (gfx1250) — matrix operand order repaired and checked by reading, not compiled, not run on hardware
//
#include <hip/hip_runtime.h>
#include <stdint.h>

// ---------------- problem constants ----------------
#define B_      2
#define S_      2048
#define HID_    2048
#define H_      16
#define NOPE_   128
#define ROPE_   64
#define VDIM_   128
#define QHD_    192
#define QLORA_  1536
#define KVLORA_ 512
#define INTER_  8192
#define RROWS   (B_*S_)          // 4096 token rows

// ---------------- WMMA types ----------------
typedef __attribute__((ext_vector_type(16))) __bf16       v16bf;
typedef __attribute__((ext_vector_type(8)))  float        v8f;
typedef __attribute__((ext_vector_type(4)))  unsigned int v4u;
typedef __attribute__((ext_vector_type(2)))  unsigned int v2u;

union Frag { v16bf bf; v4u q[2]; };

__device__ __forceinline__ unsigned short f2bf(float f) {
    unsigned u = __float_as_uint(f);
    u += 0x7FFFu + ((u >> 16) & 1u);          // round-to-nearest-even
    return (unsigned short)(u >> 16);
}
__device__ __forceinline__ float bf2f(unsigned short h) {
    return __uint_as_float(((unsigned)h) << 16);
}
__device__ __forceinline__ v8f wmma_bf16(v16bf a, v16bf b, v8f c) {
    return __builtin_amdgcn_wmma_f32_16x16x32_bf16(false, a, false, b,
                                                   (short)0, c, false, false);
}
// async copy 8 bytes global -> LDS (ASYNCcnt-tracked); ldsOff = LDS byte address
__device__ __forceinline__ void async_g2l_b64(unsigned ldsOff, const void* gp) {
    asm volatile("global_load_async_to_lds_b64 %0, %1, off"
                 :: "v"(ldsOff), "v"((unsigned long long)(uintptr_t)gp)
                 : "memory");
}
__device__ __forceinline__ void wait_async0() {
    asm volatile("s_wait_asynccnt 0x0" ::: "memory");
}

// ---------------- RMSNorm -> bf16 ----------------
__global__ __launch_bounds__(256)
void rmsnorm_bf16(const float* __restrict__ in, const float* __restrict__ w,
                  unsigned short* __restrict__ out, int cols) {
    int row = blockIdx.x;
    const float* x = in + (size_t)row * cols;
    __shared__ float red[256];
    float s = 0.f;
    for (int c = threadIdx.x; c < cols; c += 256) { float v = x[c]; s += v * v; }
    red[threadIdx.x] = s; __syncthreads();
    for (int o = 128; o > 0; o >>= 1) {
        if (threadIdx.x < o) red[threadIdx.x] += red[threadIdx.x + o];
        __syncthreads();
    }
    float inv = rsqrtf(red[0] / (float)cols + 1e-6f);
    for (int c = threadIdx.x; c < cols; c += 256)
        out[(size_t)row * cols + c] = f2bf(x[c] * inv * w[c]);
}

// ---------------- LayerNorm -> bf16 ----------------
__global__ __launch_bounds__(256)
void layernorm_bf16(const float* __restrict__ in, int ldin, int cols,
                    const float* __restrict__ w, const float* __restrict__ bias,
                    unsigned short* __restrict__ out, int ldout) {
    int row = blockIdx.x;
    const float* x = in + (size_t)row * ldin;
    __shared__ float r1[256], r2[256];
    float s = 0.f, s2 = 0.f;
    for (int c = threadIdx.x; c < cols; c += 256) { float v = x[c]; s += v; s2 += v * v; }
    r1[threadIdx.x] = s; r2[threadIdx.x] = s2; __syncthreads();
    for (int o = 128; o > 0; o >>= 1) {
        if (threadIdx.x < o) { r1[threadIdx.x] += r1[threadIdx.x + o];
                               r2[threadIdx.x] += r2[threadIdx.x + o]; }
        __syncthreads();
    }
    float mu  = r1[0] / (float)cols;
    float var = r2[0] / (float)cols - mu * mu;
    float inv = rsqrtf(var + 1e-6f);
    for (int c = threadIdx.x; c < cols; c += 256)
        out[(size_t)row * ldout + c] = f2bf((x[c] - mu) * inv * w[c] + bias[c]);
}

// ---------------- Tiled WMMA GEMM: C[4096,N] = Abf16[4096,K] * W(f32)[K,N] ----------------
// block 128x128, BK=32, 256 threads = 8 waves in a 4x2 grid, wave tile 32x64.
// A tile staged with async global->LDS DMA (ASYNCcnt); B tile converted f32->bf16 in flight.
__global__ __launch_bounds__(256)
void gemm_bf16(const unsigned short* __restrict__ A, int lda,
               const float* __restrict__ W, int ldw, int wcol0,
               unsigned short* __restrict__ outBf, float* __restrict__ outF,
               const float* __restrict__ res, int ldc, int Nlog, int K) {
    __shared__ unsigned short As[128 * 40];   // [m][k], stride 40 halves
    __shared__ unsigned short Bs[128 * 40];   // [n][k] (transposed)
    const int tid  = threadIdx.x;
    const int lane = tid & 31;
    const int wv   = tid >> 5;
    const int ln   = lane & 15;
    const int hi   = lane >> 4;
    const int wm   = wv >> 1;                 // 0..3
    const int wn   = wv & 1;                  // 0..1
    const int m0   = blockIdx.y * 128;
    const int n0   = blockIdx.x * 128;

    v8f acc[2][4];
    for (int i = 0; i < 2; ++i)
        for (int j = 0; j < 4; ++j)
            for (int r = 0; r < 8; ++r) acc[i][j][r] = 0.f;

    const int ksteps = K >> 5;
    for (int kt = 0; kt < ksteps; ++kt) {
        // stage A tile (bf16, 128x32): async DMA straight into LDS
        #pragma unroll
        for (int c = 0; c < 4; ++c) {
            int chunk = tid + c * 256;              // 0..1023
            int row = chunk >> 3;
            int kc  = (chunk & 7) << 2;
            unsigned ldsA = (unsigned)(uintptr_t)&As[row * 40 + kc];
            async_g2l_b64(ldsA, A + (size_t)(m0 + row) * lda + (kt << 5) + kc);
        }
        // stage B tile (f32 -> bf16, transposed to [n][k])
        #pragma unroll
        for (int c = 0; c < 4; ++c) {
            int chunk = tid + c * 256;
            int kr = chunk >> 5;
            int nc = (chunk & 31) << 2;
            const float* wp = W + (size_t)((kt << 5) + kr) * ldw + wcol0 + n0 + nc;
            float f0 = 0.f, f1 = 0.f, f2 = 0.f, f3 = 0.f;
            if (n0 + nc + 3 < Nlog) { f0 = wp[0]; f1 = wp[1]; f2 = wp[2]; f3 = wp[3]; }
            else {
                if (n0 + nc + 0 < Nlog) f0 = wp[0];
                if (n0 + nc + 1 < Nlog) f1 = wp[1];
                if (n0 + nc + 2 < Nlog) f2 = wp[2];
                if (n0 + nc + 3 < Nlog) f3 = wp[3];
            }
            Bs[(nc + 0) * 40 + kr] = f2bf(f0);
            Bs[(nc + 1) * 40 + kr] = f2bf(f1);
            Bs[(nc + 2) * 40 + kr] = f2bf(f2);
            Bs[(nc + 3) * 40 + kr] = f2bf(f3);
        }
        // prefetch next W tile into cache while we compute this one
        if (kt + 1 < ksteps) {
            const float* pf = W + (size_t)(((kt + 1) << 5) + (tid >> 3)) * ldw
                              + wcol0 + n0 + ((tid & 7) << 4);
            __builtin_prefetch(pf, 0, 1);
        }
        wait_async0();                 // our async A-tile DMA landed in LDS
        __syncthreads();               // everyone's tiles visible

        Frag af[2], bfr[4];
        #pragma unroll
        for (int i = 0; i < 2; ++i) {
            int m = wm * 32 + i * 16 + ln;
            af[i].q[0] = *(const v4u*)&As[m * 40 + hi * 8];
            af[i].q[1] = *(const v4u*)&As[m * 40 + 16 + hi * 8];
        }
        #pragma unroll
        for (int j = 0; j < 4; ++j) {
            int n = wn * 64 + j * 16 + ln;
            bfr[j].q[0] = *(const v4u*)&Bs[n * 40 + hi * 8];
            bfr[j].q[1] = *(const v4u*)&Bs[n * 40 + 16 + hi * 8];
        }
        #pragma unroll
        for (int i = 0; i < 2; ++i)
            #pragma unroll
            for (int j = 0; j < 4; ++j)
                acc[i][j] = wmma_bf16(af[i].bf, bfr[j].bf, acc[i][j]);
        __syncthreads();
    }

    // epilogue
    #pragma unroll
    for (int i = 0; i < 2; ++i) {
        #pragma unroll
        for (int j = 0; j < 4; ++j) {
            int row0 = m0 + wm * 32 + i * 16 + hi * 8;
            int col  = n0 + wn * 64 + j * 16 + ln;
            if (col < Nlog) {
                #pragma unroll
                for (int r = 0; r < 8; ++r) {
                    float v = acc[i][j][r];
                    size_t idx = (size_t)(row0 + r) * ldc + col;
                    if (outF) outF[idx] = v + (res ? res[idx] : 0.f);
                    else      outBf[idx] = f2bf(v);
                }
            }
        }
    }
}

// ---------------- RoPE on q_pe (in place, pairwise so no hazard) ----------------
__global__ void rope_q(unsigned short* __restrict__ q) {
    int idx = blockIdx.x * blockDim.x + threadIdx.x;
    if (idx >= RROWS * H_ * 32) return;
    int j   = idx & 31;
    int h   = (idx >> 5) & 15;
    int row = idx >> 9;
    int s   = row & (S_ - 1);
    size_t base = (size_t)row * (H_ * QHD_) + (size_t)h * QHD_ + NOPE_;
    float inv = __expf(-((float)(2 * j) / (float)ROPE_) * 9.210340371976184f); // ln(1e4)
    float ang = (float)s * inv;
    float c = cosf(ang), sn = sinf(ang);
    float x1 = bf2f(q[base + j]);
    float x2 = bf2f(q[base + j + 32]);
    q[base + j]      = f2bf(x1 * c - x2 * sn);
    q[base + j + 32] = f2bf(x2 * c + x1 * sn);
}

// ---------------- build key (k_nope ++ roped broadcast k_pe) and V^T ----------------
__global__ __launch_bounds__(256)
void kv_prep(const unsigned short* __restrict__ kv, const float* __restrict__ ckv,
             unsigned short* __restrict__ key, unsigned short* __restrict__ vT) {
    int row = blockIdx.x;                 // b*S + s
    int s = row & (S_ - 1);
    int b = row >> 11;
    __shared__ float kpe[ROPE_];
    int t = threadIdx.x;
    if (t < 32) {
        float x1 = ckv[(size_t)row * 576 + 512 + t];
        float x2 = ckv[(size_t)row * 576 + 544 + t];
        float inv = __expf(-((float)(2 * t) / (float)ROPE_) * 9.210340371976184f);
        float ang = (float)s * inv;
        float c = cosf(ang), sn = sinf(ang);
        kpe[t]      = x1 * c - x2 * sn;
        kpe[t + 32] = x2 * c + x1 * sn;
    }
    __syncthreads();
    for (int i = t; i < H_ * QHD_; i += 256) {
        int h = i / QHD_, d = i - h * QHD_;
        unsigned short v = (d < NOPE_) ? kv[(size_t)row * 4096 + h * 256 + d]
                                       : f2bf(kpe[d - NOPE_]);
        key[(size_t)row * (H_ * QHD_) + i] = v;
    }
    for (int i = t; i < H_ * VDIM_; i += 256) {
        int h = i >> 7, dv = i & 127;
        vT[(((size_t)b * H_ + h) * VDIM_ + dv) * S_ + s] =
            kv[(size_t)row * 4096 + h * 256 + 128 + dv];
    }
}

// ---------------- causal flash attention (one wave per 16 query rows) ----------------
__global__ __launch_bounds__(128)
void flash_attn(const unsigned short* __restrict__ q,
                const unsigned short* __restrict__ key,
                const unsigned short* __restrict__ vT,
                unsigned short* __restrict__ outA) {
    const int lane = threadIdx.x & 31;
    const int wv   = threadIdx.x >> 5;
    const int ln   = lane & 15;
    const int hi   = lane >> 4;
    const int h    = blockIdx.y;
    const int b    = blockIdx.z;
    const int qbase = (blockIdx.x * 4 + wv) * 16;
    __shared__ unsigned short pst[4][512];      // per-wave P staging (16x32 bf16)
    unsigned short* ps = pst[wv];

    // Q fragments (A-operand layout), held in registers for the whole kv loop
    Frag qf[6];
    const unsigned short* qrow = q + ((size_t)(b * S_ + qbase + ln) * H_ + h) * QHD_;
    #pragma unroll
    for (int f = 0; f < 6; ++f) {
        qf[f].q[0] = *(const v4u*)(qrow + f * 32 + hi * 8);
        qf[f].q[1] = *(const v4u*)(qrow + f * 32 + 16 + hi * 8);
    }

    v8f acc[8];
    for (int j = 0; j < 8; ++j) for (int r = 0; r < 8; ++r) acc[j][r] = 0.f;
    float mrow[8], lrow[8];
    for (int r = 0; r < 8; ++r) { mrow[r] = -3.0e38f; lrow[r] = 0.f; }

    const float scale = rsqrtf((float)QHD_);

    for (int kb = 0; kb < qbase + 16; kb += 32) {
        // S = Q K^T for 32 keys (two 16x16 tiles)
        v8f st[2];
        #pragma unroll
        for (int t = 0; t < 2; ++t) {
            for (int r = 0; r < 8; ++r) st[t][r] = 0.f;
            const unsigned short* krow =
                key + ((size_t)(b * S_ + kb + t * 16 + ln) * H_ + h) * QHD_;
            #pragma unroll
            for (int f = 0; f < 6; ++f) {
                Frag kf;
                kf.q[0] = *(const v4u*)(krow + f * 32 + hi * 8);
                kf.q[1] = *(const v4u*)(krow + f * 32 + 16 + hi * 8);
                st[t] = wmma_bf16(qf[f].bf, kf.bf, st[t]);
            }
        }
        // online softmax update (f32), lanes 0-15 own rows 0-7, 16-31 rows 8-15
        #pragma unroll
        for (int r = 0; r < 8; ++r) {
            int rowg = qbase + r + hi * 8;
            float s0 = st[0][r] * scale; if (kb + ln > rowg)      s0 = -3.0e38f;
            float s1 = st[1][r] * scale; if (kb + 16 + ln > rowg) s1 = -3.0e38f;
            float mx = fmaxf(s0, s1);
            mx = fmaxf(mx, __shfl_xor(mx, 1));
            mx = fmaxf(mx, __shfl_xor(mx, 2));
            mx = fmaxf(mx, __shfl_xor(mx, 4));
            mx = fmaxf(mx, __shfl_xor(mx, 8));
            float mnew  = fmaxf(mrow[r], mx);
            float alpha = __expf(mrow[r] - mnew);
            mrow[r] = mnew;
            float e0 = __expf(s0 - mnew);
            float e1 = __expf(s1 - mnew);
            float sm = e0 + e1;
            sm += __shfl_xor(sm, 1); sm += __shfl_xor(sm, 2);
            sm += __shfl_xor(sm, 4); sm += __shfl_xor(sm, 8);
            lrow[r] = lrow[r] * alpha + sm;
            #pragma unroll
            for (int j = 0; j < 8; ++j) acc[j][r] *= alpha;
            ps[(r + hi * 8) * 32 + ln]      = f2bf(e0);
            ps[(r + hi * 8) * 32 + 16 + ln] = f2bf(e1);
        }
        asm volatile("s_wait_dscnt 0" ::: "memory");   // cross-lane P via LDS
        // reload P in A-operand layout
        Frag pf;
        pf.q[0] = *(const v4u*)&ps[ln * 32 + hi * 8];
        pf.q[1] = *(const v4u*)&ps[ln * 32 + 16 + hi * 8];
        // O += P V  (V^T layout makes B-operand per-lane contiguous)
        #pragma unroll
        for (int j = 0; j < 8; ++j) {
            Frag vf;
            const unsigned short* vrow =
                vT + (((size_t)b * H_ + h) * VDIM_ + j * 16 + ln) * S_ + kb + hi * 8;
            vf.q[0] = *(const v4u*)(vrow);
            vf.q[1] = *(const v4u*)(vrow + 16);
            acc[j] = wmma_bf16(pf.bf, vf.bf, acc[j]);
        }
    }
    // write attn output [B,S,H*VDIM] bf16
    #pragma unroll
    for (int j = 0; j < 8; ++j)
        #pragma unroll
        for (int r = 0; r < 8; ++r) {
            float o = acc[j][r] / lrow[r];
            outA[(size_t)(b * S_ + qbase + r + hi * 8) * (H_ * VDIM_)
                 + h * VDIM_ + j * 16 + ln] = f2bf(o);
        }
}

// ---------------- SiLU(gate) * up ----------------
__global__ void silu_mul(const unsigned short* __restrict__ g,
                         const unsigned short* __restrict__ u,
                         unsigned short* __restrict__ o, int n) {
    int i = blockIdx.x * blockDim.x + threadIdx.x;
    if (i < n) {
        float gv = bf2f(g[i]), uv = bf2f(u[i]);
        o[i] = f2bf(gv / (1.f + __expf(-gv)) * uv);
    }
}

// ---------------- workspace arena (bytes) ----------------
static const size_t OFF_ABF   = 0;            // 16 MiB  a bf16
static const size_t OFF_QA    = 16777216;     // 24 MiB  qa f32
static const size_t OFF_QALN  = 41943040;     // 12 MiB  qa_ln bf16
static const size_t OFF_QBF   = 54525952;     // 24 MiB  q bf16 (rope'd in place)
static const size_t OFF_CKV   = 79691776;     //  9 MiB  ckv f32
static const size_t OFF_KVALN = 89128960;     //  4 MiB  kva_ln bf16
static const size_t OFF_KV    = 93323264;     // 32 MiB  kv bf16
static const size_t OFF_KEY   = 126877696;    // 24 MiB  key bf16
static const size_t OFF_VT    = 152043520;    // 16 MiB  v^T bf16
static const size_t OFF_ATTN  = 168820736;    // 16 MiB  attn bf16
static const size_t OFF_GATE  = 0;            // 64 MiB  (reuses dead phase-1 region)
static const size_t OFF_UP    = 67108864;     // 64 MiB
static const size_t OFF_MLP   = 134217728;    // 64 MiB
static const size_t OFF_M     = 201326592;    // 16 MiB  m bf16
static const size_t WS_NEEDED = 218103808;

extern "C" void kernel_launch(void* const* d_in, const int* in_sizes, int n_in,
                              void* d_out, int out_size, void* d_ws, size_t ws_size,
                              hipStream_t stream) {
    (void)in_sizes; (void)n_in; (void)out_size;
    if (ws_size < WS_NEEDED) return;

    const float* hid  = (const float*)d_in[0];
    const float* iw   = (const float*)d_in[1];
    const float* wqa  = (const float*)d_in[2];
    const float* qaw  = (const float*)d_in[3];
    const float* qab  = (const float*)d_in[4];
    const float* wqb  = (const float*)d_in[5];
    const float* wkva = (const float*)d_in[6];
    const float* kvaw = (const float*)d_in[7];
    const float* kvab = (const float*)d_in[8];
    const float* wkvb = (const float*)d_in[9];
    const float* wo   = (const float*)d_in[10];
    const float* pw   = (const float*)d_in[11];
    const float* wgu  = (const float*)d_in[12];
    const float* wd   = (const float*)d_in[13];
    float* out = (float*)d_out;
    char*  ws  = (char*)d_ws;

    unsigned short* aBf    = (unsigned short*)(ws + OFF_ABF);
    float*          qaF    = (float*)         (ws + OFF_QA);
    unsigned short* qalnBf = (unsigned short*)(ws + OFF_QALN);
    unsigned short* qBf    = (unsigned short*)(ws + OFF_QBF);
    float*          ckvF   = (float*)         (ws + OFF_CKV);
    unsigned short* kvalnBf= (unsigned short*)(ws + OFF_KVALN);
    unsigned short* kvBf   = (unsigned short*)(ws + OFF_KV);
    unsigned short* keyBf  = (unsigned short*)(ws + OFF_KEY);
    unsigned short* vtBf   = (unsigned short*)(ws + OFF_VT);
    unsigned short* attnBf = (unsigned short*)(ws + OFF_ATTN);
    unsigned short* gateBf = (unsigned short*)(ws + OFF_GATE);
    unsigned short* upBf   = (unsigned short*)(ws + OFF_UP);
    unsigned short* mlpBf  = (unsigned short*)(ws + OFF_MLP);
    unsigned short* mBf    = (unsigned short*)(ws + OFF_M);

    // a = rms_norm(h)
    rmsnorm_bf16<<<RROWS, 256, 0, stream>>>(hid, iw, aBf, HID_);
    // qa = a @ w_q_a   (f32 out, feeds layernorm)
    gemm_bf16<<<dim3(QLORA_/128, RROWS/128), 256, 0, stream>>>(
        aBf, HID_, wqa, QLORA_, 0, nullptr, qaF, nullptr, QLORA_, QLORA_, HID_);
    layernorm_bf16<<<RROWS, 256, 0, stream>>>(qaF, QLORA_, QLORA_, qaw, qab, qalnBf, QLORA_);
    // q = qa_ln @ w_q_b  (bf16)
    gemm_bf16<<<dim3((H_*QHD_)/128, RROWS/128), 256, 0, stream>>>(
        qalnBf, QLORA_, wqb, H_*QHD_, 0, qBf, nullptr, nullptr, H_*QHD_, H_*QHD_, QLORA_);
    // ckv = a @ w_kv_a  (N=576, f32)
    gemm_bf16<<<dim3(5, RROWS/128), 256, 0, stream>>>(
        aBf, HID_, wkva, KVLORA_+ROPE_, 0, nullptr, ckvF, nullptr,
        KVLORA_+ROPE_, KVLORA_+ROPE_, HID_);
    layernorm_bf16<<<RROWS, 256, 0, stream>>>(ckvF, KVLORA_+ROPE_, KVLORA_, kvaw, kvab,
                                              kvalnBf, KVLORA_);
    // kv = kva_ln @ w_kv_b  (bf16)
    gemm_bf16<<<dim3((H_*(NOPE_+VDIM_))/128, RROWS/128), 256, 0, stream>>>(
        kvalnBf, KVLORA_, wkvb, H_*(NOPE_+VDIM_), 0, kvBf, nullptr, nullptr,
        H_*(NOPE_+VDIM_), H_*(NOPE_+VDIM_), KVLORA_);
    // rope q_pe in place; assemble key and V^T
    rope_q<<<(RROWS*H_*32 + 255)/256, 256, 0, stream>>>(qBf);
    kv_prep<<<RROWS, 256, 0, stream>>>(kvBf, ckvF, keyBf, vtBf);
    // causal flash attention
    flash_attn<<<dim3(S_/64, H_, B_), 128, 0, stream>>>(qBf, keyBf, vtBf, attnBf);
    // h_mid = h + attn @ w_o  (f32, into d_out)
    gemm_bf16<<<dim3(HID_/128, RROWS/128), 256, 0, stream>>>(
        attnBf, H_*VDIM_, wo, HID_, 0, nullptr, out, hid, HID_, HID_, H_*VDIM_);
    // m = rms_norm(h_mid)
    rmsnorm_bf16<<<RROWS, 256, 0, stream>>>(out, pw, mBf, HID_);
    // gate / up halves of w_gate_up
    gemm_bf16<<<dim3(INTER_/128, RROWS/128), 256, 0, stream>>>(
        mBf, HID_, wgu, 2*INTER_, 0,      gateBf, nullptr, nullptr, INTER_, INTER_, HID_);
    gemm_bf16<<<dim3(INTER_/128, RROWS/128), 256, 0, stream>>>(
        mBf, HID_, wgu, 2*INTER_, INTER_, upBf,   nullptr, nullptr, INTER_, INTER_, HID_);
    silu_mul<<<(RROWS*INTER_ + 255)/256, 256, 0, stream>>>(gateBf, upBf, mlpBf, RROWS*INTER_);
    // out = h_mid + mlp @ w_down (residual accumulate into d_out)
    gemm_bf16<<<dim3(HID_/128, RROWS/128), 256, 0, stream>>>(
        mlpBf, INTER_, wd, HID_, 0, nullptr, out, out, HID_, HID_, INTER_);
}
